// JointDescriptorLoss_45586782880293
// MI455X (gfx1250) — matrix operand order repaired and checked by reading, not compiled, not run on hardware
//
#include <hip/hip_runtime.h>

// Problem constants (match the JAX reference)
#define BB  4
#define CC  256
#define HC  60
#define WC  80
#define NN  (HC * WC)          // 4800
#define M_POS 1.0f
#define M_NEG 0.2f
#define THR2  64.0f            // THRESHOLD^2

// Workgroup macro-tile: 192(n) x 96(m), 8 waves in a 4(n) x 2(m) grid,
// each wave computes a 48x48 sub-tile as 3x3 WMMA 16x16 tiles, K in 8 steps of 32.
#define WGN 192
#define WGM 96
#define ROWB 80                // padded LDS row stride (64B of K data + 16B pad)
#define ABUF (WGN * ROWB)      // 15360 B per A buffer
#define BBUF (WGM * ROWB)      //  7680 B per B buffer

typedef __attribute__((ext_vector_type(16))) _Float16 v16h;
typedef __attribute__((ext_vector_type(8)))  float    v8f;
typedef __attribute__((ext_vector_type(4)))  unsigned int v4u;

__global__ void zero_out_kernel(float* out) { out[0] = 0.0f; }

static __device__ __forceinline__ unsigned int pack2h(float a, float b) {
    union { _Float16 h[2]; unsigned int u; } u;
    u.h[0] = (_Float16)a;
    u.h[1] = (_Float16)b;
    return u.u;
}

// ---- CDNA5 async global->LDS copies (ASYNCcnt path), inline asm -----------
static __device__ __forceinline__ void async_copy_b128(unsigned lds_off, const void* gaddr) {
    asm volatile("global_load_async_to_lds_b128 %0, %1, off"
                 :: "v"(lds_off), "v"(gaddr) : "memory");
}
static __device__ __forceinline__ void async_copy_b64(unsigned lds_off, const void* gaddr) {
    asm volatile("global_load_async_to_lds_b64 %0, %1, off"
                 :: "v"(lds_off), "v"(gaddr) : "memory");
}
// Each iteration issues exactly 6 async copies per thread; waiting to <=6
// outstanding guarantees the PREVIOUS iteration's copies have completed
// (async loads complete in order) while this iteration's are still in flight.
static __device__ __forceinline__ void wait_async_le6() {
    asm volatile("s_wait_asynccnt 0x6" ::: "memory");
}

// ===========================================================================
// Pass 1: f32 [C][N] -> f16 [N][C] transpose+convert (LDS-tiled, coalesced)
// ===========================================================================
__global__ __launch_bounds__(256)
void transpose_convert_kernel(const float* __restrict__ inA,
                              const float* __restrict__ inB,
                              _Float16* __restrict__ outA,
                              _Float16* __restrict__ outB)
{
    __shared__ float tile[32][33];
    const int which = blockIdx.z >> 2;            // 0 = desc_a, 1 = desc_b
    const int b     = blockIdx.z & 3;
    const float*    src = (which ? inB : inA) + (size_t)b * CC * NN;
    _Float16*       dst = (which ? outB : outA) + (size_t)b * NN * CC;
    const int n0 = blockIdx.x * 32;
    const int c0 = blockIdx.y * 32;
#pragma unroll
    for (int r = threadIdx.y; r < 32; r += 8)
        tile[r][threadIdx.x] = src[(size_t)(c0 + r) * NN + n0 + threadIdx.x];
    __syncthreads();
#pragma unroll
    for (int r = threadIdx.y; r < 32; r += 8)
        dst[(size_t)(n0 + r) * CC + c0 + threadIdx.x] = (_Float16)tile[threadIdx.x][r];
}

// ===========================================================================
// Shared epilogue helper: hinge loss + wave reduction + atomic accumulate
// ===========================================================================
static __device__ __forceinline__ void epilogue_reduce(
    const v8f acc[3][3], const float* s_wx, const float* s_wy,
    const float cbx[3], const float cby[3],
    int wr, int hf, int lane, float* out)
{
    float lsum = 0.0f;
#pragma unroll
    for (int i = 0; i < 3; ++i) {
#pragma unroll
        for (int r = 0; r < 8; ++r) {
            const int   nloc = wr * 48 + i * 16 + r + (hf << 3);
            const float wx = s_wx[nloc];
            const float wy = s_wy[nloc];
#pragma unroll
            for (int j = 0; j < 3; ++j) {
                const float dx  = wx - cbx[j];
                const float dy  = wy - cby[j];
                const float d2  = dx * dx + dy * dy;
                const float sim = acc[i][j][r];
                const float pos = fmaxf(M_POS - sim, 0.0f);
                const float neg = fmaxf(sim - M_NEG, 0.0f);
                lsum += (d2 <= THR2) ? pos : neg;
            }
        }
    }
#pragma unroll
    for (int off = 16; off > 0; off >>= 1)
        lsum += __shfl_down(lsum, off, 32);
    if (lane == 0) {
        const float scale = 1.0f / ((float)BB * (float)NN * (float)NN);
        atomicAdd(out, lsum * scale);
    }
}

static __device__ __forceinline__ void warp_centers_setup(
    const float* Hb, int tid, int n0, float* s_wx, float* s_wy)
{
    if (tid < WGN) {
        const int n  = n0 + tid;
        const int gi = n / WC;
        const int gj = n - gi * WC;
        const float x = (float)gj * 8.0f + 4.0f;   // centers are (x,y)=(j,i)*8+4
        const float y = (float)gi * 8.0f + 4.0f;
        const float wz  = Hb[6] * x + Hb[7] * y + Hb[8];
        const float inv = 1.0f / wz;
        s_wx[tid] = (Hb[0] * x + Hb[1] * y + Hb[2]) * inv;
        s_wy[tid] = (Hb[3] * x + Hb[4] * y + Hb[5]) * inv;
    }
}

// ===========================================================================
// Pass 2 (fast path): f16 [N][C] panels, double-buffered async copy to LDS,
// WMMA main loop with prefetch pipelining on ASYNCcnt.
// ===========================================================================
__global__ __launch_bounds__(256)
void gemm_loss_f16t_kernel(const _Float16* __restrict__ A16,
                           const _Float16* __restrict__ B16,
                           const float* __restrict__ Hmat,
                           float* __restrict__ out)
{
    const int tid  = threadIdx.x;
    const int lane = tid & 31;
    const int w    = tid >> 5;
    const int wr   = w >> 1;            // wave row (n) 0..3
    const int wc   = w & 1;             // wave col (m) 0..1
    const int hf   = lane >> 4;
    const int l    = lane & 15;

    const int m0 = blockIdx.x * WGM;    // 50
    const int n0 = blockIdx.y * WGN;    // 25
    const int b  = blockIdx.z;          // 4

    const char* __restrict__ Ab = (const char*)(A16 + (size_t)b * NN * CC);  // [N][C] f16
    const char* __restrict__ Bb = (const char*)(B16 + (size_t)b * NN * CC);  // [N][C] f16

    __shared__ __align__(16) unsigned char s_a[2 * ABUF];   // 30720 B
    __shared__ __align__(16) unsigned char s_b[2 * BBUF];   // 15360 B
    __shared__ float s_wx[WGN];
    __shared__ float s_wy[WGN];

    warp_centers_setup(Hmat + b * 9, tid, n0, s_wx, s_wy);

    float cbx[3], cby[3];
#pragma unroll
    for (int j = 0; j < 3; ++j) {
        const int m  = m0 + (wc * 3 + j) * 16 + l;
        const int gi = m / WC;
        const int gj = m - gi * WC;
        cbx[j] = (float)gj * 8.0f + 4.0f;
        cby[j] = (float)gi * 8.0f + 4.0f;
    }

    const unsigned lds_a = (unsigned)(uintptr_t)&s_a[0];
    const unsigned lds_b = (unsigned)(uintptr_t)&s_b[0];

    // Per-thread static chunk assignment (uniform: 3 b128 + 3 b64 = 6 asyncs).
    // A panel: 192 rows x 64 B = 768 x 16B chunks.
    // B panel:  96 rows x 64 B = 768 x  8B chunks.
    int  a_row[3], a_part[3], b_row[3], b_part[3];
#pragma unroll
    for (int t = 0; t < 3; ++t) {
        const int ca = tid + t * 256;
        a_row[t]  = ca >> 2;
        a_part[t] = ca & 3;
        const int cb = tid + t * 256;
        b_row[t]  = cb >> 3;
        b_part[t] = cb & 7;
    }

    // ---- issue helper: 6 async copies for K-step kp into buffer `buf` ----
    auto issue_panels = [&](int buf, int kp) {
#pragma unroll
        for (int t = 0; t < 3; ++t) {
            const char* g = Ab + ((size_t)(n0 + a_row[t]) * CC + kp) * 2 + a_part[t] * 16;
            async_copy_b128(lds_a + buf * ABUF + a_row[t] * ROWB + a_part[t] * 16, g);
        }
#pragma unroll
        for (int t = 0; t < 3; ++t) {
            const char* g = Bb + ((size_t)(m0 + b_row[t]) * CC + kp) * 2 + b_part[t] * 8;
            async_copy_b64(lds_b + buf * BBUF + b_row[t] * ROWB + b_part[t] * 8, g);
        }
    };

    v8f acc[3][3] = {};

    // Prologue: fill buffer 0 with K-step 0.
    issue_panels(0, 0);

#pragma unroll 1
    for (int t = 0; t < CC / 32; ++t) {
        const int kk  = t * 32;
        const int cur = t & 1;
        // Prefetch next K-step into the other buffer (wraps on last iter;
        // the wrapped data is never consumed, it just keeps addresses legal).
        issue_panels(cur ^ 1, (kk + 32) & (CC - 1));

        wait_async_le6();      // previous K-step's 6 copies have landed
        __syncthreads();       // ... in every wave of the workgroup

        // ---- consume: 12 ds_load_b128 + 9 WMMAs --------------------------
        union Frag { v16h h; v4u q[2]; };
        Frag af[3], bf[3];
        const unsigned char* pa = s_a + cur * ABUF;
        const unsigned char* pb = s_b + cur * BBUF;
#pragma unroll
        for (int i = 0; i < 3; ++i) {
            const int row = (wr * 3 + i) * 16 + l;
            af[i].q[0] = *(const v4u*)(pa + row * ROWB + hf * 16);        // K 0-7 / 8-15
            af[i].q[1] = *(const v4u*)(pa + row * ROWB + 32 + hf * 16);   // K 16-23 / 24-31
        }
#pragma unroll
        for (int j = 0; j < 3; ++j) {
            const int row = (wc * 3 + j) * 16 + l;
            bf[j].q[0] = *(const v4u*)(pb + row * ROWB + hf * 16);        // K 0-7 / 8-15
            bf[j].q[1] = *(const v4u*)(pb + row * ROWB + 32 + hf * 16);   // K 16-23 / 24-31
        }

#pragma unroll
        for (int i = 0; i < 3; ++i)
#pragma unroll
            for (int j = 0; j < 3; ++j)
                acc[i][j] = __builtin_amdgcn_wmma_f32_16x16x32_f16(
                    false, af[i].h, false, bf[j].h,
                    (short)0, acc[i][j], false, false);

        __syncthreads();   // all reads of `cur` done before it is refilled
    }

    epilogue_reduce(acc, s_wx, s_wy, cbx, cby, wr, hf, lane, out);
}

// ===========================================================================
// Fallback (ws too small): f32 loads + cvt + LDS fragment images
// ===========================================================================
__global__ __launch_bounds__(256)
void gemm_loss_f32_kernel(const float* __restrict__ desc_a,
                          const float* __restrict__ desc_b,
                          const float* __restrict__ Hmat,
                          float* __restrict__ out)
{
    const int tid  = threadIdx.x;
    const int lane = tid & 31;
    const int w    = tid >> 5;
    const int wr   = w >> 1;
    const int wc   = w & 1;
    const int hf   = lane >> 4;
    const int l    = lane & 15;

    const int m0 = blockIdx.x * WGM;
    const int n0 = blockIdx.y * WGN;
    const int b  = blockIdx.z;

    const float* __restrict__ A  = desc_a + (size_t)b * CC * NN;
    const float* __restrict__ Bd = desc_b + (size_t)b * CC * NN;

    __shared__ __align__(16) unsigned int s_a[12 * 256];
    __shared__ __align__(16) unsigned int s_b[6 * 256];
    __shared__ float s_wx[WGN];
    __shared__ float s_wy[WGN];

    warp_centers_setup(Hmat + b * 9, tid, n0, s_wx, s_wy);

    float cbx[3], cby[3];
#pragma unroll
    for (int j = 0; j < 3; ++j) {
        const int m  = m0 + (wc * 3 + j) * 16 + l;
        const int gi = m / WC;
        const int gj = m - gi * WC;
        cbx[j] = (float)gj * 8.0f + 4.0f;
        cby[j] = (float)gi * 8.0f + 4.0f;
    }

    v8f acc[3][3] = {};
    const v4u* pa = (const v4u*)s_a;
    const v4u* pb = (const v4u*)s_b;

#pragma unroll 1
    for (int kk = 0; kk < CC; kk += 32) {
#pragma unroll
        for (int it = 0; it < 12; ++it) {
            const int idx = tid + it * 256;
            const int lf  = idx & 31;
            const int v   = (idx >> 5) & 7;
            const int ti  = idx >> 8;
            const int hfw = lf >> 4;
            const int ll  = lf & 15;
            const int k = kk + ((v & 3) << 1) + (hfw << 3) + ((v >> 2) << 4);
            const int n = n0 + ti * 16 + ll;
            const float f0 = A[(size_t)k * NN + n];
            const float f1 = A[(size_t)(k + 1) * NN + n];
            s_a[ti * 256 + ((v >> 2) << 7) + lf * 4 + (v & 3)] = pack2h(f0, f1);
        }
#pragma unroll
        for (int it = 0; it < 6; ++it) {
            const int idx = tid + it * 256;
            const int lf  = idx & 31;
            const int v   = (idx >> 5) & 7;
            const int ti  = idx >> 8;
            const int hfw = lf >> 4;
            const int ll  = lf & 15;
            const int k = kk + ((v & 3) << 1) + (hfw << 3) + ((v >> 2) << 4);
            const int m = m0 + ti * 16 + ll;
            const float f0 = Bd[(size_t)k * NN + m];
            const float f1 = Bd[(size_t)(k + 1) * NN + m];
            s_b[ti * 256 + ((v >> 2) << 7) + lf * 4 + (v & 3)] = pack2h(f0, f1);
        }
        __syncthreads();

        union Frag { v16h h; v4u q[2]; };
        Frag af[3], bf[3];
#pragma unroll
        for (int i = 0; i < 3; ++i) {
            const int t = (wr * 3 + i) * 64;
            af[i].q[0] = pa[t + lane];
            af[i].q[1] = pa[t + 32 + lane];
        }
#pragma unroll
        for (int j = 0; j < 3; ++j) {
            const int t = (wc * 3 + j) * 64;
            bf[j].q[0] = pb[t + lane];
            bf[j].q[1] = pb[t + 32 + lane];
        }
#pragma unroll
        for (int i = 0; i < 3; ++i)
#pragma unroll
            for (int j = 0; j < 3; ++j)
                acc[i][j] = __builtin_amdgcn_wmma_f32_16x16x32_f16(
                    false, af[i].h, false, bf[j].h,
                    (short)0, acc[i][j], false, false);
        __syncthreads();
    }

    epilogue_reduce(acc, s_wx, s_wy, cbx, cby, wr, hf, lane, out);
}

// ===========================================================================
extern "C" void kernel_launch(void* const* d_in, const int* in_sizes, int n_in,
                              void* d_out, int out_size, void* d_ws, size_t ws_size,
                              hipStream_t stream)
{
    const float* desc_a = (const float*)d_in[0];   // [B,C,Hc,Wc] f32
    const float* desc_b = (const float*)d_in[1];   // [B,C,Hc,Wc] f32
    const float* Hmat   = (const float*)d_in[2];   // [B,3,3]     f32
    // d_in[3] = valid_mask : unused by the reference computation.
    float* out = (float*)d_out;

    zero_out_kernel<<<1, 1, 0, stream>>>(out);

    const size_t one  = (size_t)BB * NN * CC * sizeof(_Float16);
    const size_t need = 2 * one;

    dim3 grid(NN / WGM, NN / WGN, BB);   // (50, 25, 4)
    dim3 block(256);

    if (ws_size >= need) {
        _Float16* A16 = (_Float16*)d_ws;
        _Float16* B16 = (_Float16*)((char*)d_ws + one);
        dim3 tg(NN / 32, CC / 32, 2 * BB);  // (150, 8, 8)
        transpose_convert_kernel<<<tg, dim3(32, 8), 0, stream>>>(desc_a, desc_b, A16, B16);
        gemm_loss_f16t_kernel<<<grid, block, 0, stream>>>(A16, B16, Hmat, out);
    } else {
        gemm_loss_f32_kernel<<<grid, block, 0, stream>>>(desc_a, desc_b, Hmat, out);
    }
}
